// FeatureExtract_22892175688331
// MI455X (gfx1250) — matrix operand order repaired and checked by reading, not compiled, not run on hardware
//
#include <hip/hip_runtime.h>

typedef __attribute__((ext_vector_type(16))) _Float16 v16h;
typedef __attribute__((ext_vector_type(4)))  _Float16 v4h;
typedef __attribute__((ext_vector_type(8)))  float    v8f;

// ---------------------------------------------------------------------------
// Fragment-ready LDS layout (inverse of CDNA5 ISA 7.12.2 VGPR layouts, wave32):
//  A (16x32 f16):  element (m,k) -> lane = m | ((k>>3)&1)<<4,
//                                   slot = (k&7) | ((k>>4)&1)<<3
//  B (32x16 f16):  element (k,n) -> lane = n | ((k>>3)&1)<<4, slot = (k&7) | ((k>>4)&1)<<3
//  C/D f32 16x16:  lane&15 = N, row m = r + 8*(lane>>4)
// Each lane's 16 operand halves are contiguous -> fragment load = 2x ds_load_b128.
// ---------------------------------------------------------------------------

// ---------------------------------------------------------------------------
// Generic GEMM:  Y(O x NC) = W(O x Cin) * X(Cin x NC) + bias[o]
// Block = 256 thr = 8 waves (2 o-subtiles x 4 col-subtiles), tile 32 x 64, K-step 32.
// GUARD=true: element-wise staging with zero padding (only used when Cin%32!=0).
// ---------------------------------------------------------------------------
template <bool GUARD>
__global__ __launch_bounds__(256) void gemm_wmma(
    const float* __restrict__ W, const float* __restrict__ bias,
    const float* __restrict__ X, float* __restrict__ Y,
    int O, int Cin, int NC) {
  __shared__ __align__(32) _Float16 Asf[2 * 32 * 16];  // 2 o-subtiles
  __shared__ __align__(32) _Float16 Bsf[4 * 32 * 16];  // 4 col-subtiles
  int tid = threadIdx.x, lane = tid & 31, wid = tid >> 5;
  int wm = wid & 1, wn = wid >> 1;
  int o0 = blockIdx.y * 32;
  int c0 = blockIdx.x * 64;
  v8f acc = {};
  // fast-path A staging: one float4 of W per thread
  int ar = tid >> 3;                    // tile row 0..31
  int ac = (tid & 7) * 4;               // k offset 0..28
  _Float16* aDst = &Asf[(((ar >> 4) * 32 + ((ar & 15) | (((ac >> 3) & 1) << 4))) * 16)
                        + ((ac & 7) | (((ac >> 4) & 1) << 3))];
  for (int k0 = 0; k0 < Cin; k0 += 32) {
    if (GUARD) {
      for (int i = tid; i < 32 * 32; i += 256) {
        int r = i >> 5, c = i & 31;
        int k = k0 + c;
        float v = (k < Cin) ? W[(size_t)(o0 + r) * Cin + k] : 0.f;
        Asf[((r >> 4) * 32 + ((r & 15) | (((c >> 3) & 1) << 4))) * 16
            + ((c & 7) | (((c >> 4) & 1) << 3))] = (_Float16)v;
      }
      for (int i = tid; i < 32 * 64; i += 256) {
        int r = i >> 6, c = i & 63;
        int k = k0 + r;
        float v = (k < Cin) ? X[(size_t)k * NC + c0 + c] : 0.f;
        Bsf[((c >> 4) * 32 + ((((r >> 3) & 1) << 4) | (c & 15))) * 16 + ((r & 7) | (((r >> 4) & 1) << 3))] = (_Float16)v;
      }
    } else {
      const float4 wa = *(const float4*)&W[(size_t)(o0 + ar) * Cin + k0 + ac];
      v4h pa = { (_Float16)wa.x, (_Float16)wa.y, (_Float16)wa.z, (_Float16)wa.w };
      *(v4h*)aDst = pa;
#pragma unroll
      for (int h = 0; h < 2; ++h) {
        int q = tid + h * 256;          // quad 0..511 of the 32x64 B tile
        int r = q >> 4;                 // k row 0..31
        int cq = (q & 15) * 4;          // col 0..60
        const float4 xb = *(const float4*)&X[(size_t)(k0 + r) * NC + c0 + cq];
        _Float16* bd = &Bsf[((cq >> 4) * 32 + ((((r >> 3) & 1) << 4) | (cq & 15))) * 16 + ((r & 7) | (((r >> 4) & 1) << 3))];
        bd[0]  = (_Float16)xb.x;        // cols cq..cq+3 -> lanes n..n+3 (stride 16 halves)
        bd[16] = (_Float16)xb.y;
        bd[32] = (_Float16)xb.z;
        bd[48] = (_Float16)xb.w;
      }
    }
    __syncthreads();
    if (k0 + 32 < Cin)
      __builtin_prefetch(&X[(size_t)(k0 + 32) * NC + c0 + lane], 0, 0);
    v16h a = *(const v16h*)&Asf[(wm * 32 + lane) * 16];
    v16h b = *(const v16h*)&Bsf[(wn * 32 + lane) * 16];
    acc = __builtin_amdgcn_wmma_f32_16x16x32_f16(false, a, false, b,
                                                 (short)0, acc, false, false);
    __syncthreads();
  }
  int n = lane & 15, hf = lane >> 4;
  int col = c0 + wn * 16 + n;
#pragma unroll
  for (int r = 0; r < 8; ++r) {
    int o = o0 + wm * 16 + r + 8 * hf;
    float v = acc[r] + (bias ? bias[o] : 0.f);
    Y[(size_t)o * NC + col] = v;
  }
}

// ---------------------------------------------------------------------------
// EdgeConv GEMM.  y[o, (n,k)] = sum_c2 W[o,c2] * E[c2,(n,k)]
//   E[c2] = F[c2, idx[n,k]] - F[c2, n]   (c2 <  Cf)
//         = F[c2-Cf, n]                  (c2 >= Cf)
// PASS 1: accumulate groupnorm sum/sumsq per (b, group) via atomics (no store).
// PASS 2: recompute, normalize, leaky-relu, max over k (16 cols == one point),
//         optional outer relu, store F_out (O x B*N).
// Grid: (NK/64, O/32, B).  N=2048, K=16, cols = n*16 + k.
// ---------------------------------------------------------------------------
template <int PASS>
__global__ __launch_bounds__(256) void edge_wmma(
    const float* __restrict__ W, const float* __restrict__ F,
    const int* __restrict__ idx, const float* __restrict__ gamma,
    const float* __restrict__ beta, float* __restrict__ stats,
    float* __restrict__ Fout, int Cf, int O, int relu_out) {
  const int BN = 4096, N = 2048;
  const int C2 = 2 * Cf;
  __shared__ __align__(32) _Float16 Asf[2 * 32 * 16];
  __shared__ __align__(32) _Float16 Bsf[4 * 32 * 16];
  int tid = threadIdx.x, lane = tid & 31, wid = tid >> 5;
  int wm = wid & 1, wn = wid >> 1;
  int o0 = blockIdx.y * 32;
  int colBase = blockIdx.x * 64;
  int b = blockIdx.z;
  int bN = b * N;
  const int* idxb = idx + (size_t)bN * 16;
  int ar = tid >> 3;
  int ac = (tid & 7) * 4;
  _Float16* aDst = &Asf[(((ar >> 4) * 32 + ((ar & 15) | (((ac >> 3) & 1) << 4))) * 16)
                        + ((ac & 7) | (((ac >> 4) & 1) << 3))];
  v8f acc = {};
  for (int k0 = 0; k0 < C2; k0 += 32) {
    const float4 wa = *(const float4*)&W[(size_t)(o0 + ar) * C2 + k0 + ac];
    v4h pa = { (_Float16)wa.x, (_Float16)wa.y, (_Float16)wa.z, (_Float16)wa.w };
    *(v4h*)aDst = pa;
    for (int i = tid; i < 32 * 64; i += 256) {
      int r = i >> 6, c = i & 63;
      int c2 = k0 + r;
      int colg = colBase + c;
      int n = colg >> 4, kk = colg & 15;
      float v;
      if (c2 < Cf) {
        int j = idxb[(n << 4) + kk];
        v = F[(size_t)c2 * BN + bN + j] - F[(size_t)c2 * BN + bN + n];
      } else {
        v = F[(size_t)(c2 - Cf) * BN + bN + n];
      }
      Bsf[((c >> 4) * 32 + ((((r >> 3) & 1) << 4) | (c & 15))) * 16 + ((r & 7) | (((r >> 4) & 1) << 3))] = (_Float16)v;
    }
    __syncthreads();
    v16h a = *(const v16h*)&Asf[(wm * 32 + lane) * 16];
    v16h bb = *(const v16h*)&Bsf[(wn * 32 + lane) * 16];
    acc = __builtin_amdgcn_wmma_f32_16x16x32_f16(false, a, false, bb,
                                                 (short)0, acc, false, false);
    __syncthreads();
  }
  int g = o0 / (O >> 2);  // group size O/4 >= 32, o-tile never straddles
  if (PASS == 1) {
    float s = 0.f, q = 0.f;
#pragma unroll
    for (int r = 0; r < 8; ++r) { s += acc[r]; q += acc[r] * acc[r]; }
#pragma unroll
    for (int m = 16; m >= 1; m >>= 1) {
      s += __shfl_xor(s, m, 32);
      q += __shfl_xor(q, m, 32);
    }
    if (lane == 0) {
      atomicAdd(&stats[(b * 4 + g) * 2 + 0], s);
      atomicAdd(&stats[(b * 4 + g) * 2 + 1], q);
    }
  } else {
    float cnt = (float)((O >> 2) * N * 16);
    float mu = stats[(b * 4 + g) * 2 + 0] / cnt;
    float var = stats[(b * 4 + g) * 2 + 1] / cnt - mu * mu;
    float rstd = rsqrtf(var + 1e-5f);
    int hf = lane >> 4;
    float mx[8];
#pragma unroll
    for (int r = 0; r < 8; ++r) {
      int o = o0 + wm * 16 + r + 8 * hf;
      float yn = (acc[r] - mu) * rstd * gamma[o] + beta[o];
      float lv = yn > 0.f ? yn : 0.2f * yn;
#pragma unroll
      for (int m = 1; m <= 8; m <<= 1)
        lv = fmaxf(lv, __shfl_xor(lv, m, 32));  // k-max within 16-lane half
      mx[r] = lv;
    }
    if ((lane & 15) == 0) {
      int npt = blockIdx.x * 4 + wn;  // the point this 16-col tile covers
#pragma unroll
      for (int r = 0; r < 8; ++r) {
        int o = o0 + wm * 16 + r + 8 * hf;
        float v = relu_out ? fmaxf(mx[r], 0.f) : mx[r];
        Fout[(size_t)o * BN + bN + npt] = v;
      }
    }
  }
}

// ---------------------------------------------------------------------------
// Farthest point sampling: one block per batch; sequential m-1 argmax steps.
// ---------------------------------------------------------------------------
__global__ __launch_bounds__(256) void fps_kernel(const float* __restrict__ x,
                                                  int* __restrict__ out,
                                                  int m, int N) {
  int b = blockIdx.x, tid = threadIdx.x;
  __shared__ float px[2048], py[2048], pz[2048], dd[2048];
  __shared__ float rb[256];
  __shared__ int ri[256];
  __shared__ int lastS;
  for (int n = tid; n < N; n += 256) {
    px[n] = x[(size_t)b * 3 * N + 0 * N + n];
    py[n] = x[(size_t)b * 3 * N + 1 * N + n];
    pz[n] = x[(size_t)b * 3 * N + 2 * N + n];
    dd[n] = 1e10f;
  }
  if (tid == 0) { out[(size_t)b * m] = 0; lastS = 0; }
  __syncthreads();
  for (int t = 1; t < m; ++t) {
    int last = lastS;
    float lx = px[last], ly = py[last], lz = pz[last];
    float bestv = -1e30f; int besti = 0;
    for (int n = tid; n < N; n += 256) {
      float dx = px[n] - lx, dy = py[n] - ly, dz = pz[n] - lz;
      float dn = fminf(dd[n], dx * dx + dy * dy + dz * dz);
      dd[n] = dn;
      if (dn > bestv) { bestv = dn; besti = n; }
    }
    rb[tid] = bestv; ri[tid] = besti;
    __syncthreads();
    for (int s = 128; s > 0; s >>= 1) {
      if (tid < s) {
        if (rb[tid + s] > rb[tid] ||
            (rb[tid + s] == rb[tid] && ri[tid + s] < ri[tid])) {
          rb[tid] = rb[tid + s]; ri[tid] = ri[tid + s];
        }
      }
      __syncthreads();
    }
    if (tid == 0) { out[(size_t)b * m + t] = ri[0]; lastS = ri[0]; }
    __syncthreads();
  }
}

// ---------------------------------------------------------------------------
// Squared norms per point, then kNN (top-16 smallest distance), wave per query.
// ---------------------------------------------------------------------------
__global__ void sqnorm_kernel(const float* __restrict__ F, float* __restrict__ sq,
                              int Cf, int BN) {
  int q = blockIdx.x * 256 + threadIdx.x;
  if (q >= BN) return;
  float s = 0.f;
  for (int c = 0; c < Cf; ++c) { float v = F[(size_t)c * BN + q]; s += v * v; }
  sq[q] = s;
}

__global__ __launch_bounds__(256) void knn_kernel(const float* __restrict__ F,
                                                  const float* __restrict__ sq,
                                                  int* __restrict__ idx,
                                                  int Cf, int N, int BN) {
  __shared__ float sd[8][512];
  __shared__ int si[8][512];
  int wid = threadIdx.x >> 5, lane = threadIdx.x & 31;
  int q = blockIdx.x * 8 + wid;  // grid sized so q < BN always
  int b = q / N, i = q % N;
  int bN = b * N;
  float bd[16]; int bi[16];
#pragma unroll
  for (int t = 0; t < 16; ++t) { bd[t] = 1e30f; bi[t] = 0; }
  float sqi = sq[q];
  for (int j = lane; j < N; j += 32) {
    float dot = 0.f;
    for (int c = 0; c < Cf; ++c)
      dot += F[(size_t)c * BN + bN + i] * F[(size_t)c * BN + bN + j];
    float d = sqi + sq[bN + j] - 2.f * dot;
    if (d < bd[15]) {
      int t = 15;
      while (t > 0 && bd[t - 1] > d) { bd[t] = bd[t - 1]; bi[t] = bi[t - 1]; --t; }
      bd[t] = d; bi[t] = j;
    }
  }
  for (int t = 0; t < 16; ++t) { sd[wid][lane * 16 + t] = bd[t]; si[wid][lane * 16 + t] = bi[t]; }
  __syncthreads();
  if (lane == 0) {
    for (int t = 0; t < 16; ++t) {
      float best = 1e31f; int bj = 0, bp = 0;
      for (int p = 0; p < 512; ++p) {
        float v = sd[wid][p];
        if (v < best) { best = v; bj = si[wid][p]; bp = p; }
      }
      sd[wid][bp] = 1e32f;
      idx[(size_t)q * 16 + t] = bj;
    }
  }
}

// ---------------------------------------------------------------------------
// Normalization helpers. Activations: 0=none, 1=relu, 2=leaky(0.2).
// ---------------------------------------------------------------------------
__global__ void gn_stats(const float* __restrict__ Y, float* __restrict__ stats,
                         int rowsPerGroup, int NC, int N) {
  __shared__ float ss[256], sq2[256];
  int b = blockIdx.x, g = blockIdx.y, G = gridDim.y;
  float s = 0.f, q = 0.f;
  int tot = rowsPerGroup * N;
  for (int i = threadIdx.x; i < tot; i += 256) {
    int r = g * rowsPerGroup + i / N;
    int c = b * N + i % N;
    float v = Y[(size_t)r * NC + c];
    s += v; q += v * v;
  }
  ss[threadIdx.x] = s; sq2[threadIdx.x] = q;
  __syncthreads();
  for (int st = 128; st > 0; st >>= 1) {
    if (threadIdx.x < st) { ss[threadIdx.x] += ss[threadIdx.x + st]; sq2[threadIdx.x] += sq2[threadIdx.x + st]; }
    __syncthreads();
  }
  if (threadIdx.x == 0) {
    float mu = ss[0] / tot;
    float var = sq2[0] / tot - mu * mu;
    stats[(b * G + g) * 2 + 0] = mu;
    stats[(b * G + g) * 2 + 1] = rsqrtf(var + 1e-5f);
  }
}

__global__ void gn_apply(float* __restrict__ Y, const float* __restrict__ stats,
                         const float* __restrict__ gamma, const float* __restrict__ beta,
                         int rowsPerGroup, int groups, int NC, int N, int act, int O) {
  int t = blockIdx.x * 256 + threadIdx.x;
  if (t >= O * NC) return;
  int r = t / NC, c = t % NC;
  int b = c / N;
  int g = r / rowsPerGroup;
  float mu = stats[(b * groups + g) * 2], rs = stats[(b * groups + g) * 2 + 1];
  float v = (Y[t] - mu) * rs * gamma[r] + beta[r];
  if (act == 1) v = fmaxf(v, 0.f);
  else if (act == 2) v = v > 0.f ? v : 0.2f * v;
  Y[t] = v;
}

__global__ void bn_stats(const float* __restrict__ Y, float* __restrict__ stats, int NC) {
  __shared__ float ss[256], sq2[256];
  int r = blockIdx.x;
  float s = 0.f, q = 0.f;
  for (int c = threadIdx.x; c < NC; c += 256) {
    float v = Y[(size_t)r * NC + c];
    s += v; q += v * v;
  }
  ss[threadIdx.x] = s; sq2[threadIdx.x] = q;
  __syncthreads();
  for (int st = 128; st > 0; st >>= 1) {
    if (threadIdx.x < st) { ss[threadIdx.x] += ss[threadIdx.x + st]; sq2[threadIdx.x] += sq2[threadIdx.x + st]; }
    __syncthreads();
  }
  if (threadIdx.x == 0) {
    float mu = ss[0] / NC;
    float var = sq2[0] / NC - mu * mu;
    stats[r * 2 + 0] = mu;
    stats[r * 2 + 1] = rsqrtf(var + 1e-5f);
  }
}

__global__ void bn_apply(float* __restrict__ Y, const float* __restrict__ stats,
                         const float* __restrict__ g, const float* __restrict__ be,
                         int NC, int O) {
  int t = blockIdx.x * 256 + threadIdx.x;
  if (t >= O * NC) return;
  int r = t / NC;
  float v = (Y[t] - stats[r * 2]) * stats[r * 2 + 1] * g[r] + be[r];
  Y[t] = fmaxf(v, 0.f);
}

__global__ void colmax(const float* __restrict__ Y, float* __restrict__ out,
                       int O, int NC, int N, int rowOff) {
  int t = blockIdx.x * 256 + threadIdx.x;
  if (t >= 2 * O) return;
  int b = t / O, r = t % O;
  float m = -1e30f;
  for (int n = 0; n < N; ++n) m = fmaxf(m, Y[(size_t)r * NC + b * N + n]);
  out[b * 1920 + rowOff + r] = m;
}

// ---------------------------------------------------------------------------
// Small data-movement kernels.
// ---------------------------------------------------------------------------
__global__ void x_to_rows(const float* __restrict__ x, float* __restrict__ X3, int N) {
  int t = blockIdx.x * 256 + threadIdx.x;
  if (t >= 3 * 2 * N) return;
  int c = t / (2 * N), col = t % (2 * N);
  int b = col / N, n = col % N;
  X3[(size_t)c * 2 * N + col] = x[(size_t)b * 3 * N + c * N + n];
}

__global__ void gather_pts(const float* __restrict__ x, const int* __restrict__ fidx,
                           float* __restrict__ X3, int Np, int N) {
  int t = blockIdx.x * 256 + threadIdx.x;
  if (t >= 3 * 2 * Np) return;
  int c = t / (2 * Np), col = t % (2 * Np);
  int b = col / Np, i = col % Np;
  X3[(size_t)c * 2 * Np + col] = x[(size_t)b * 3 * N + c * N + fidx[b * Np + i]];
}

// ---------------------------------------------------------------------------
// Final head: gf = relu(groupnorm1(Wg . [gf0 gf1 gf2] + bg)) -> gfeat (B,1920)
// ---------------------------------------------------------------------------
__global__ void head_kernel(const float* __restrict__ gf0, const float* __restrict__ gf1,
                            const float* __restrict__ gf2, const float* __restrict__ Wg,
                            const float* __restrict__ bg, const float* __restrict__ gg,
                            const float* __restrict__ beg, float* __restrict__ gfeat,
                            float* __restrict__ outg) {
  __shared__ float ss[256], sq2[256];
  int b = blockIdx.x, tid = threadIdx.x;
  float w0 = Wg[0], w1 = Wg[1], w2 = Wg[2], bb = bg[0];
  float s = 0.f, q = 0.f;
  for (int t = tid; t < 1920; t += 256) {
    float v = w0 * gf0[b * 1920 + t] + w1 * gf1[b * 1920 + t] + w2 * gf2[b * 1920 + t] + bb;
    gfeat[b * 1920 + t] = v;
    s += v; q += v * v;
  }
  ss[tid] = s; sq2[tid] = q;
  __syncthreads();
  for (int st = 128; st > 0; st >>= 1) {
    if (tid < st) { ss[tid] += ss[tid + st]; sq2[tid] += sq2[tid + st]; }
    __syncthreads();
  }
  float mu = ss[0] / 1920.f;
  float rs = rsqrtf(sq2[0] / 1920.f - mu * mu + 1e-5f);
  float G = gg[0], BE = beg[0];
  for (int t = tid; t < 1920; t += 256) {
    float v = gfeat[b * 1920 + t];
    v = fmaxf((v - mu) * rs * G + BE, 0.f);
    gfeat[b * 1920 + t] = v;
    outg[b * 1920 + t] = v;
  }
}

// code = relu(Wc(192x15) @ gfeat.reshape(B,15,128) + bc) -> (B,192,128)
__global__ void code_kernel(const float* __restrict__ gfeat, const float* __restrict__ Wc,
                            const float* __restrict__ bc, float* __restrict__ out) {
  int t = blockIdx.x * 256 + threadIdx.x;
  if (t >= 2 * 192 * 128) return;
  int b = t / (192 * 128);
  int rem = t % (192 * 128);
  int o = rem / 128, j = rem % 128;
  float s = bc[o];
#pragma unroll
  for (int c = 0; c < 15; ++c) s += Wc[o * 15 + c] * gfeat[b * 1920 + c * 128 + j];
  out[t] = fmaxf(s, 0.f);
}

// ---------------------------------------------------------------------------
// Host orchestration.
// Input flattening order (setup_inputs dict order, nested dicts in insertion
// order): x, W0,b0,g0,be0, (Wl,gl,bel)x4, Wm1,bm1,gm,bem,Wm2,bm2,
// cl1{W1,b1..W6,b6,g1,be1..g6,be6}, cl2{...}, Wg,bg,gg,beg,Wc,bc.
// ---------------------------------------------------------------------------
extern "C" void kernel_launch(void* const* d_in, const int* in_sizes, int n_in,
                              void* d_out, int out_size, void* d_ws, size_t ws_size,
                              hipStream_t stream) {
  (void)in_sizes; (void)n_in; (void)out_size; (void)ws_size;
  const float* x = (const float*)d_in[0];
#define PF(i) ((const float*)d_in[(i)])
  float* out = (float*)d_out;
  float* ws = (float*)d_ws;
  const size_t BN = 4096;

  float* f0   = ws;                          // 64   x 4096
  float* fg   = f0 + (size_t)64 * BN;        // 1920 x 4096  (f1..f4 row segments)
  float* h1   = fg + (size_t)1920 * BN;      // 1920 x 4096
  float* h2   = h1 + (size_t)1920 * BN;      // 1920 x 4096
  float* sqb  = h2 + (size_t)1920 * BN;      // 4096
  int*   idxb = (int*)(sqb + 4096);          // 65536 ints
  float* est  = (float*)(idxb + 65536);      // 16 edge-gn accumulators
  float* gst  = est + 16;                    // 2048 norm stats
  int*   fi1  = (int*)(gst + 2048);          // 1024 ints
  int*   fi2  = fi1 + 1024;                  // 512 ints
  float* X30  = (float*)(fi2 + 512);         // 3 x 4096
  float* X31  = X30 + 12288;                 // 3 x 1024
  float* X32  = X31 + 3072;                  // 3 x 512
  float* c1b  = X32 + 1536;                  // 2,097,152 (conv chain 1)
  float* c2b  = c1b + 2097152;               // 1,048,576 (conv chain 2)
  float* gf0b = c2b + 1048576;               // 2 x 1920
  float* gf1b = gf0b + 3840;
  float* gf2b = gf1b + 3840;
  float* gfe  = gf2b + 3840;                 // 2 x 1920

  // ---- FPS sampling + sampled point gathers ----
  fps_kernel<<<2, 256, 0, stream>>>(x, fi1, 512, 2048);
  fps_kernel<<<2, 256, 0, stream>>>(x, fi2, 256, 2048);
  gather_pts<<<12, 256, 0, stream>>>(x, fi1, X31, 512, 2048);
  gather_pts<<<6, 256, 0, stream>>>(x, fi2, X32, 256, 2048);

  // ---- f0 = relu(gn4(W0 x + b0)) ----
  x_to_rows<<<48, 256, 0, stream>>>(x, X30, 2048);
  gemm_wmma<true><<<dim3(64, 2), 256, 0, stream>>>(PF(1), PF(2), X30, f0, 64, 3, 4096);
  gn_stats<<<dim3(2, 4), 256, 0, stream>>>(f0, gst, 16, 4096, 2048);
  gn_apply<<<(64 * 4096) / 256, 256, 0, stream>>>(f0, gst, PF(3), PF(4), 16, 4, 4096, 2048, 1, 64);

  // ---- 4 EdgeConv blocks, writing into fg row segments ----
  struct ECfg { const float* F; float* Fo; int Cf, wi, gi, bi, relu; };
  ECfg ec[4] = {
    { f0,                      fg,                      64,  5,  6,  7,  1 },
    { fg,                      fg + (size_t)128 * BN,   128, 8,  9,  10, 1 },
    { fg + (size_t)128 * BN,   fg + (size_t)384 * BN,   256, 11, 12, 13, 0 },
    { fg + (size_t)384 * BN,   fg + (size_t)896 * BN,   512, 14, 15, 16, 0 },
  };
  for (int l = 0; l < 4; ++l) {
    int Cf = ec[l].Cf, O = 2 * Cf;
    sqnorm_kernel<<<16, 256, 0, stream>>>(ec[l].F, sqb, Cf, 4096);
    knn_kernel<<<512, 256, 0, stream>>>(ec[l].F, sqb, idxb, Cf, 2048, 4096);
    hipMemsetAsync(est, 0, 16 * sizeof(float), stream);
    edge_wmma<1><<<dim3(512, O / 32, 2), 256, 0, stream>>>(
        PF(ec[l].wi), ec[l].F, idxb, nullptr, nullptr, est, nullptr, Cf, O, 0);
    edge_wmma<2><<<dim3(512, O / 32, 2), 256, 0, stream>>>(
        PF(ec[l].wi), ec[l].F, idxb, PF(ec[l].gi), PF(ec[l].bi), est, ec[l].Fo, Cf, O, ec[l].relu);
  }

  // ---- Main 1920x1920 branch: h1 = Wm1 fg + bm1; gn32+leaky; h2 = Wm2 h1 + bm2; max->gf0 ----
  gemm_wmma<false><<<dim3(64, 60), 256, 0, stream>>>(PF(17), PF(18), fg, h1, 1920, 1920, 4096);
  gn_stats<<<dim3(2, 32), 256, 0, stream>>>(h1, gst, 60, 4096, 2048);
  gn_apply<<<(1920 * 4096) / 256, 256, 0, stream>>>(h1, gst, PF(19), PF(20), 60, 32, 4096, 2048, 2, 1920);
  gemm_wmma<false><<<dim3(64, 60), 256, 0, stream>>>(PF(21), PF(22), h1, h2, 1920, 1920, 4096);
  colmax<<<15, 256, 0, stream>>>(h2, gf0b, 1920, 4096, 2048, 0);

  // ---- PointNet conv chains on sampled sets ----
  auto convchain = [&](const float* X3, float* base, int Np, int pbase, float* gfout) {
    int NC = 2 * Np;
    float* a64   = base;
    float* b64   = a64 + (size_t)64 * NC;
    float* x128  = b64 + (size_t)64 * NC;
    float* x256  = x128 + (size_t)128 * NC;
    float* x512  = x256 + (size_t)256 * NC;
    float* x1024 = x512 + (size_t)512 * NC;
    const float* ins[6] = { X3, a64, b64, x128, x256, x512 };
    float* outs[6]      = { a64, b64, x128, x256, x512, x1024 };
    int Os[6] = { 64, 64, 128, 256, 512, 1024 };
    int Cs[6] = { 3, 64, 64, 128, 256, 512 };
    for (int j = 0; j < 6; ++j) {
      if (Cs[j] % 32)
        gemm_wmma<true><<<dim3(NC / 64, Os[j] / 32), 256, 0, stream>>>(
            PF(pbase + 2 * j), PF(pbase + 2 * j + 1), ins[j], outs[j], Os[j], Cs[j], NC);
      else
        gemm_wmma<false><<<dim3(NC / 64, Os[j] / 32), 256, 0, stream>>>(
            PF(pbase + 2 * j), PF(pbase + 2 * j + 1), ins[j], outs[j], Os[j], Cs[j], NC);
      bn_stats<<<Os[j], 256, 0, stream>>>(outs[j], gst, NC);
      bn_apply<<<(Os[j] * NC) / 256, 256, 0, stream>>>(
          outs[j], gst, PF(pbase + 12 + 2 * j), PF(pbase + 12 + 2 * j + 1), NC, Os[j]);
    }
    colmax<<<(2 * 1024 + 255) / 256, 256, 0, stream>>>(x1024, gfout, 1024, NC, Np, 0);
    colmax<<<(2 * 512 + 255) / 256, 256, 0, stream>>>(x512, gfout, 512, NC, Np, 1024);
    colmax<<<(2 * 256 + 255) / 256, 256, 0, stream>>>(x256, gfout, 256, NC, Np, 1536);
    colmax<<<(2 * 128 + 255) / 256, 256, 0, stream>>>(x128, gfout, 128, NC, Np, 1792);
  };
  convchain(X31, c1b, 512, 23, gf1b);
  convchain(X32, c2b, 256, 47, gf2b);

  // ---- Head: gfeat + code ----
  head_kernel<<<2, 256, 0, stream>>>(gf0b, gf1b, gf2b, PF(71), PF(72), PF(73), PF(74),
                                     gfe, out + 49152);
  code_kernel<<<192, 256, 0, stream>>>(gfe, PF(75), PF(76), out);
#undef PF
}
